// MultiHeadAttention_8572754722984
// MI455X (gfx1250) — matrix operand order repaired and checked by reading, not compiled, not run on hardware
//
#include <hip/hip_runtime.h>
#include <hip/hip_bf16.h>

// MHA forward for B=4, S=2048, D=1024, H=16, DK=DV=64 on gfx1250 (wave32).
// Stage 1: QKV projections via f16 WMMA -> f16 [B,H,S,64] in d_ws.
// Stage 2: streaming attention (raw exp, multiplicative key mask) via f16 WMMA,
//          with K-chunk staging done by the Tensor Data Mover when available.

typedef _Float16 hf;
typedef _Float16 v16h __attribute__((ext_vector_type(16)));
typedef _Float16 v8h  __attribute__((ext_vector_type(8)));
typedef float    v8f  __attribute__((ext_vector_type(8)));
typedef unsigned int u32x4 __attribute__((ext_vector_type(4)));
typedef int          i32x4 __attribute__((ext_vector_type(4)));
typedef int          i32x8 __attribute__((ext_vector_type(8)));

union V16 { v16h v; v8h h[2]; };

struct __attribute__((aligned(8))) H4 { hf a, b, c, d; };

#define WMMA_F16(A, B, C) \
  __builtin_amdgcn_wmma_f32_16x16x32_f16(false, (A), false, (B), (short)0, (C), false, false)

#if defined(__has_builtin)
#if __has_builtin(__builtin_amdgcn_tensor_load_to_lds) && \
    __has_builtin(__builtin_amdgcn_s_wait_tensorcnt)
#define USE_TDM 1
#endif
#endif

static constexpr int Bb = 4, Ss = 2048, Dd = 1024, Hh = 16, DK = 64;

// ---------------------------------------------------------------------------
// Projection: Out[b,h,s,d] = (X[b,s,:] @ W[:, h*64+d] + bias) * scale, f16 out.
// Block: 256 threads (8 waves). Tile: M=128 (rows of B*S), N=64 (one head), K=32.
// ---------------------------------------------------------------------------
__global__ __launch_bounds__(256) void proj_wmma_kernel(
    const float* __restrict__ X, const float* __restrict__ W,
    const float* __restrict__ bias, hf* __restrict__ Out, float scale)
{
  __shared__ __attribute__((aligned(16))) hf sA[128 * 40];  // [m][k], stride 40 halves
  __shared__ __attribute__((aligned(16))) hf sB[64 * 40];   // [n][k] transposed

  const int t    = threadIdx.x;
  const int wv   = t >> 5;
  const int lane = t & 31;
  const int g    = lane >> 4;
  const int n    = lane & 15;
  const int m0   = blockIdx.x * 128;
  const int n0   = blockIdx.y * 64;

  v8f acc[4];
  const v8f z8 = {0.f, 0.f, 0.f, 0.f, 0.f, 0.f, 0.f, 0.f};
#pragma unroll
  for (int i = 0; i < 4; ++i) acc[i] = z8;

  for (int k0 = 0; k0 < Dd; k0 += 32) {
    __syncthreads();
    // Stage A tile: 128x32 f32 -> f16, coalesced float4 reads.
#pragma unroll
    for (int i = 0; i < 4; ++i) {
      int gid = t + 256 * i;               // 0..1023 float4 chunks
      int row = gid >> 3;
      int kq  = (gid & 7) << 2;
      float4 xv = *reinterpret_cast<const float4*>(
          &X[(size_t)(m0 + row) * Dd + k0 + kq]);
      H4 hv{(hf)xv.x, (hf)xv.y, (hf)xv.z, (hf)xv.w};
      *reinterpret_cast<H4*>(&sA[row * 40 + kq]) = hv;
    }
    // Stage B tile: 32x64 f32 -> f16 transposed into sB[n][k].
#pragma unroll
    for (int i = 0; i < 8; ++i) {
      int gid = t + 256 * i;               // 0..2047
      int nn  = gid & 63;
      int kk  = gid >> 6;
      sB[nn * 40 + kk] = (hf)W[(size_t)(k0 + kk) * Dd + n0 + nn];
    }
    __syncthreads();

    // A fragment: lane row = wv*16 + n; K elements 16g..16g+15 contiguous.
    V16 ua;
    const hf* arow = &sA[(wv * 16 + n) * 40];
    ua.h[0] = *reinterpret_cast<const v8h*>(arow + 16 * g);
    ua.h[1] = *reinterpret_cast<const v8h*>(arow + 16 * g + 8);

#pragma unroll
    for (int nt = 0; nt < 4; ++nt) {
      // B fragment: lane column = nt*16 + n; K elements 16g..16g+15 contiguous.
      V16 ub;
      const hf* brow = &sB[(nt * 16 + n) * 40 + 16 * g];
      ub.h[0] = *reinterpret_cast<const v8h*>(brow);
      ub.h[1] = *reinterpret_cast<const v8h*>(brow + 8);
      acc[nt] = WMMA_F16(ua.v, ub.v, acc[nt]);
    }
  }

  // Epilogue: D fragment row = r + 8g, column = nt*16 + n. Head-major f16 out.
  const int bidx  = m0 >> 11;        // m0 / 2048 (block never spans batches)
  const int sbase = m0 & 2047;
  const int h     = n0 >> 6;         // one head per block column tile
#pragma unroll
  for (int nt = 0; nt < 4; ++nt) {
    int col  = n0 + nt * 16 + n;
    float bv = bias[col];
    int d    = col & 63;
#pragma unroll
    for (int r = 0; r < 8; ++r) {
      int s = sbase + wv * 16 + r + 8 * g;
      float val = (acc[nt][r] + bv) * scale;
      Out[(((size_t)bidx * Hh + h) * Ss + s) * DK + d] = (hf)val;
    }
  }
}

// ---------------------------------------------------------------------------
// Attention: per (b,h), 64 query rows per block (4 waves x 16 rows), stream
// keys in chunks of 32. scores -> exp -> *mask[key] -> rowsum via ones-WMMA,
// context accumulated with WMMA; final divide by (rowsum + 1e-8).
// q is pre-scaled by 1/sqrt(DK) in the projection.
// K chunk staging: TDM (tensor_load_to_lds) when available, overlapped with
// the V-transpose staging; TENSORcnt wait + barrier publishes LDS.
// ---------------------------------------------------------------------------
__global__ __launch_bounds__(128) void attn_wmma_kernel(
    const hf* __restrict__ Qh, const hf* __restrict__ Kh,
    const hf* __restrict__ Vh, const float* __restrict__ mask,
    float* __restrict__ Out)
{
  __shared__ __attribute__((aligned(16))) hf sK[32 * 72];    // [key][d] stride 72 halves (144 B)
  __shared__ __attribute__((aligned(16))) hf sV[64 * 40];    // [dv][key] stride 40 (transposed)
  __shared__ __attribute__((aligned(16))) hf sP[4][16 * 40]; // wave-private P tiles

  const int t    = threadIdx.x;
  const int wv   = t >> 5;
  const int lane = t & 31;
  const int g    = lane >> 4;
  const int n    = lane & 15;
  const int qt   = blockIdx.x;
  const int bh   = blockIdx.y;
  const int b    = bh >> 4;
  const int h    = bh & 15;

  const size_t head_off = (size_t)bh * Ss * DK;
  const int qrow0 = qt * 64 + wv * 16;

  // Persistent A fragments for q (16 rows x 64 d): two K=32 chunks.
  V16 aq[2];
  {
    const hf* qrow = Qh + head_off + (size_t)(qrow0 + n) * DK;
#pragma unroll
    for (int c = 0; c < 2; ++c) {
      aq[c].h[0] = *reinterpret_cast<const v8h*>(qrow + 32 * c + 16 * g);
      aq[c].h[1] = *reinterpret_cast<const v8h*>(qrow + 32 * c + 16 * g + 8);
    }
  }
  V16 ones;
#pragma unroll
  for (int i = 0; i < 16; ++i) ones.v[i] = (hf)1.0f;

  const v8f z8 = {0.f, 0.f, 0.f, 0.f, 0.f, 0.f, 0.f, 0.f};
  v8f cacc[4];
#pragma unroll
  for (int i = 0; i < 4; ++i) cacc[i] = z8;
  v8f racc = z8;

  const float* mrow = mask + (size_t)b * Ss;

#ifdef USE_TDM
  // TDM D# pieces that are loop-invariant.
  // Group1: data_size=2B (1), pad_enable=1, pad_interval=32 dwords (code 4),
  // pad_amount=4 dwords (code 3); tensor_dim0=64, tensor_dim1=32,
  // tile_dim0=64, tile_dim1=32, dim0_stride=64.
  const i32x8 tdm_g1 = {0x07110000, 0x00400000, 0x00200000, 0x00400000,
                        32, 64, 0, 0};
  const i32x4 tdm_zero4 = {0, 0, 0, 0};
  const i32x8 tdm_zero8 = {0, 0, 0, 0, 0, 0, 0, 0};
  const unsigned lds_k_base = (unsigned)(uintptr_t)&sK[0];
#endif

  for (int kc = 0; kc < Ss; kc += 32) {
    __syncthreads();

#ifdef USE_TDM
    if (wv == 0) {
      // Issue DMA of the 32x64 f16 K chunk into sK (with 16B row padding).
      unsigned long long ga =
          (unsigned long long)(uintptr_t)(Kh + head_off + (size_t)kc * DK);
      u32x4 g0 = {1u,                       // count=1 (valid descriptor)
                  lds_k_base,               // lds_addr
                  (unsigned)ga,             // global_addr[31:0]
                  (unsigned)((ga >> 32) & 0x01FFFFFFu) | 0x80000000u}; // type=2
      __builtin_amdgcn_tensor_load_to_lds(g0, tdm_g1, tdm_zero4, tdm_zero4,
                                          tdm_zero8, 0);
    }
#else
    // Stage K chunk [32 keys][64 d] (pure copy, b128).
#pragma unroll
    for (int i = 0; i < 2; ++i) {
      int cid = t + 128 * i;                 // 0..255 v8h chunks
      int row = cid >> 3;
      int dq  = (cid & 7) << 3;
      *reinterpret_cast<v8h*>(&sK[row * 72 + dq]) =
          *reinterpret_cast<const v8h*>(Kh + head_off + (size_t)(kc + row) * DK + dq);
    }
#endif

    // Stage V chunk transposed -> sV[dv][key] (overlaps the TDM transfer).
#pragma unroll
    for (int i = 0; i < 4; ++i) {
      int cid = t + 128 * i;                 // 0..511 4-half chunks
      int row = cid >> 4;
      int dv4 = (cid & 15) << 2;
      H4 hv = *reinterpret_cast<const H4*>(
          Vh + head_off + (size_t)(kc + row) * DK + dv4);
      sV[(dv4 + 0) * 40 + row] = hv.a;
      sV[(dv4 + 1) * 40 + row] = hv.b;
      sV[(dv4 + 2) * 40 + row] = hv.c;
      sV[(dv4 + 3) * 40 + row] = hv.d;
    }

#ifdef USE_TDM
    if (wv == 0) __builtin_amdgcn_s_wait_tensorcnt(0);
#endif
    __syncthreads();

    // Scores: two 16-key subtiles, K=64 contraction each (2 WMMA).
#pragma unroll
    for (int kt = 0; kt < 2; ++kt) {
      v8f s = z8;
#pragma unroll
      for (int c = 0; c < 2; ++c) {
        V16 ub;   // B = K^T fragment: lane column = key, K elements = d.
        const hf* brow = &sK[(kt * 16 + n) * 72 + 32 * c + 16 * g];
        ub.h[0] = *reinterpret_cast<const v8h*>(brow);
        ub.h[1] = *reinterpret_cast<const v8h*>(brow + 8);
        s = WMMA_F16(aq[c].v, ub.v, s);
      }
      float mv = mrow[kc + kt * 16 + n];     // mask over key dim
#pragma unroll
      for (int r = 0; r < 8; ++r) {
        float e = __expf(s[r]) * mv;         // raw exp, faithful to reference
        sP[wv][(r + 8 * g) * 40 + kt * 16 + n] = (hf)e;
      }
    }

    // Re-fragment P (16x32) as an A operand; wave-private LDS, DS in-order.
    V16 ap;
    const hf* prow = &sP[wv][n * 40];
    ap.h[0] = *reinterpret_cast<const v8h*>(prow + 16 * g);
    ap.h[1] = *reinterpret_cast<const v8h*>(prow + 16 * g + 8);

    // Row-sums: P @ ones -> every column holds the row sum (matches cacc rows).
    racc = WMMA_F16(ap.v, ones.v, racc);

    // Context: P (16x32) @ V (32x64).
#pragma unroll
    for (int nt = 0; nt < 4; ++nt) {
      V16 ub;
      const hf* vrow = &sV[(nt * 16 + n) * 40 + 16 * g];
      ub.h[0] = *reinterpret_cast<const v8h*>(vrow);
      ub.h[1] = *reinterpret_cast<const v8h*>(vrow + 8);
      cacc[nt] = WMMA_F16(ap.v, ub.v, cacc[nt]);
    }
  }

  // Epilogue: out[b, s, h*64 + dv] fp32.
#pragma unroll
  for (int nt = 0; nt < 4; ++nt) {
#pragma unroll
    for (int r = 0; r < 8; ++r) {
      int srow = qrow0 + r + 8 * g;
      int col  = h * 64 + nt * 16 + n;
      Out[((size_t)(b * Ss + srow)) * (Hh * DK) + col] =
          cacc[nt][r] / (racc[r] + 1e-8f);
    }
  }
}

extern "C" void kernel_launch(void* const* d_in, const int* in_sizes, int n_in,
                              void* d_out, int out_size, void* d_ws, size_t ws_size,
                              hipStream_t stream) {
  const float* Q    = (const float*)d_in[0];
  const float* K    = (const float*)d_in[1];
  const float* V    = (const float*)d_in[2];
  const float* mask = (const float*)d_in[3];
  const float* Wq   = (const float*)d_in[4];
  const float* bq   = (const float*)d_in[5];
  const float* Wk   = (const float*)d_in[6];
  const float* bk   = (const float*)d_in[7];
  const float* Wv   = (const float*)d_in[8];
  const float* bv   = (const float*)d_in[9];
  float* out = (float*)d_out;

  const size_t head_elems = (size_t)Bb * Hh * Ss * DK;   // 8,388,608 f16 each
  hf* qh = (hf*)d_ws;
  hf* kh = qh + head_elems;
  hf* vh = kh + head_elems;

  dim3 gp(64, 16), bp(256);
  // Fold 1/sqrt(DK)=0.125 into the q projection.
  proj_wmma_kernel<<<gp, bp, 0, stream>>>(Q, Wq, bq, qh, 0.125f);
  proj_wmma_kernel<<<gp, bp, 0, stream>>>(K, Wk, bk, kh, 1.0f);
  proj_wmma_kernel<<<gp, bp, 0, stream>>>(V, Wv, bv, vh, 1.0f);

  dim3 ga(Ss / 64, Bb * Hh), ba(128);
  attn_wmma_kernel<<<ga, ba, 0, stream>>>(qh, kh, vh, mask, out);
}
